// RoPEAttention_20040317403202
// MI455X (gfx1250) — matrix operand order repaired and checked by reading, not compiled, not run on hardware
//
#include <hip/hip_runtime.h>
#include <hip/hip_bf16.h>

typedef _Float16 half8  __attribute__((ext_vector_type(8)));
typedef _Float16 half16 __attribute__((ext_vector_type(16)));
typedef float    f32x8  __attribute__((ext_vector_type(8)));
typedef float    f32x2  __attribute__((ext_vector_type(2)));
typedef unsigned int u32x4 __attribute__((ext_vector_type(4)));
typedef int      i32x8  __attribute__((ext_vector_type(8)));
typedef int      i32x4  __attribute__((ext_vector_type(4)));

#define NQ    4096
#define NK    8192
#define HEADS 4
#define HD    64
#define IDIM  256

// log2(10000)/8  (freqs[i] = 10000^(-i/8) = exp2(-i * L2T8))
#define L2T8     1.6609640474436811f
// (1/sqrt(64)) * log2(e): fold softmax base-2 conversion into q scale
#define QSCALE   0.18033688011112043f

__device__ __forceinline__ f32x8 wmma_f16(half16 a, half16 b, f32x8 c) {
  return __builtin_amdgcn_wmma_f32_16x16x32_f16(false, a, false, b, (short)0, c,
                                                false, false);
}

// A-matrix (16x32 f16, MxK) fragment: lane holds row M=lane&15.
// lanes 0-15: K = j ; lanes 16-31: K = 16+j
__device__ __forceinline__ half16 load_a16(const _Float16* rowbase, int lane) {
  const int kb0 = (lane & 16) ? 16 : 0;
  half8 lo = *reinterpret_cast<const half8*>(rowbase + kb0);
  half8 hi = *reinterpret_cast<const half8*>(rowbase + kb0 + 8);
  half16 r;
#pragma unroll
  for (int j = 0; j < 8; ++j) { r[j] = lo[j]; r[j + 8] = hi[j]; }
  return r;
}

// B-matrix (32x16 f16, KxN) fragment: lane holds col N=lane&15,
// K = (lane<16?0:16)+j  -> 16 contiguous halves, pointer pre-adjusted.
__device__ __forceinline__ half16 load_b16(const _Float16* p) {
  half8 lo = *reinterpret_cast<const half8*>(p);
  half8 hi = *reinterpret_cast<const half8*>(p + 8);
  half16 r;
#pragma unroll
  for (int j = 0; j < 8; ++j) { r[j] = lo[j]; r[j + 8] = hi[j]; }
  return r;
}

// ---------------------------------------------------------------------------
// Tensor Data Mover: 2-D tile load Global->LDS (D# per CDNA5 ISA).
// data_size = 2 bytes. Dims/strides in elements. Issued once per wave.
// 6-arg builtin variant: (g0, g1, g2, g3, g_extra, cpol).
// ---------------------------------------------------------------------------
__device__ __forceinline__ void tdm_load_2d_f16(unsigned lds_addr,
                                                const void* gptr,
                                                unsigned tile_x, unsigned tile_y,
                                                unsigned tensor_x,
                                                unsigned tensor_y,
                                                unsigned stride_x) {
  unsigned long long ga = (unsigned long long)gptr;
  u32x4 g0;
  g0[0] = 1u;                                   // count=1, user mode
  g0[1] = lds_addr;                             // lds_addr (bytes)
  g0[2] = (unsigned)(ga & 0xffffffffu);         // global_addr[31:0]
  g0[3] = (unsigned)((ga >> 32) & 0x01ffffffu)  // global_addr[56:32]
          | (2u << 30);                         // type = 2 (image)
  i32x8 g1;
  g1[0] = (int)(1u << 16);                      // data_size=1 (2B), mask=0
  g1[1] = (int)((tensor_x & 0xffffu) << 16);            // tensor_dim0[15:0]
  g1[2] = (int)(((tensor_x >> 16) & 0xffffu) |
                ((tensor_y & 0xffffu) << 16));          // dim0 hi | dim1 lo
  g1[3] = (int)(((tensor_y >> 16) & 0xffffu) |
                ((tile_x & 0xffffu) << 16));            // dim1 hi | tile_dim0
  g1[4] = (int)(tile_y & 0xffffu);                      // tile_dim1, tile_dim2=0
  g1[5] = (int)stride_x;                                // tensor_dim0_stride lo
  g1[6] = 0;                                            // stride0 hi | stride1 lo
  g1[7] = 0;
  i32x4 z4 = {};
  i32x8 z8 = {};
  __builtin_amdgcn_tensor_load_to_lds(g0, g1, z4, z4, z8, 0);
}

__device__ __forceinline__ unsigned lds_off(const void* p) {
  return (unsigned)(unsigned long long)p;  // low 32 bits = LDS address
}

// ---------------------------------------------------------------------------
// Generic projection GEMM: C(M x 256) = X(M x K) @ W(K x 256) + bias.
// f32 WMMA 16x16x4. One wave per 16x16 tile; 4 waves/block; grid = (M/16)*4.
// mode 0: Q proj -> rope+scale -> f16 qh[h][n][d]
// mode 1: K proj -> rope       -> f16 kh[h][n][d]
// mode 2: V proj ->            -> f16 vT[h][d][n]
// mode 3: O proj ->            -> f32 out[n][256]
// ---------------------------------------------------------------------------
__global__ __launch_bounds__(128) void proj_kernel(
    const float* __restrict__ X, const float* __restrict__ W,
    const float* __restrict__ bias, int M, int K, int mode,
    _Float16* __restrict__ outh, float* __restrict__ outf,
    const int* __restrict__ nke_ptr) {
  const int wave = threadIdx.x >> 5;
  const int lane = threadIdx.x & 31;
  const int lrow = lane & 15;
  const bool hi = lane >= 16;

  const int mtile = blockIdx.x >> 2;
  const int ntile = (blockIdx.x & 3) * 4 + wave;
  const int row = mtile * 16 + lrow;   // A row for this lane
  const int col = ntile * 16 + lrow;   // B col for this lane
  const int ko = hi ? 2 : 0;

  f32x8 c = {};
  const float* xrow = X + (size_t)row * K + ko;
  const float* wcol = W + (size_t)ko * IDIM + col;
  for (int kk = 0; kk < K; kk += 4) {
    f32x2 a = *reinterpret_cast<const f32x2*>(xrow + kk);
    f32x2 b;
    b.x = wcol[(size_t)kk * IDIM];
    b.y = wcol[(size_t)kk * IDIM + IDIM];
    c = __builtin_amdgcn_wmma_f32_16x16x4_f32(false, a, false, b, (short)0, c,
                                              false, false);
  }

  const int nke = nke_ptr ? nke_ptr[0] : 0;
#pragma unroll
  for (int r = 0; r < 8; ++r) {
    const int orow = mtile * 16 + r + (hi ? 8 : 0);
    float v = c[r] + bias[col];
    if (mode <= 1) {  // rope modes (Q / K)
      const int hh = col >> 6, d = col & 63;
      const float partner = __shfl_xor(v, 1, 32);  // value at d^1 (same row)
      const bool dorope = (mode == 0) || (orow < M - nke);
      if (dorope) {
        const int pos = (mode == 0) ? orow : (orow & (NQ - 1));
        const int i = d >> 1;
        const float t = (i < 16) ? (float)(pos & 63) : (float)(pos >> 6);
        const int fi = (i < 16) ? i : (i - 16);
        const float freq = exp2f(-(float)fi * L2T8);
        const float ang = t * freq;
        const float cs = cosf(ang), sn = sinf(ang);
        const float rot = (d & 1) ? partner : -partner;
        v = v * cs + rot * sn;
      }
      if (mode == 0) {
        v *= QSCALE;
        outh[((size_t)hh * NQ + orow) * HD + d] = (_Float16)v;
      } else {
        outh[((size_t)hh * NK + orow) * HD + d] = (_Float16)v;
      }
    } else if (mode == 2) {  // V transposed
      const int hh = col >> 6, d = col & 63;
      outh[((size_t)hh * HD + d) * NK + orow] = (_Float16)v;
    } else {  // output projection, f32
      outf[(size_t)orow * IDIM + col] = v;
    }
  }
}

// ---------------------------------------------------------------------------
// Flash attention: block = 4 waves sharing one head; wave = 16 queries.
// K/V tiles staged in LDS by the Tensor Data Mover (double-buffered),
// scores and P*V via v_wmma_f32_16x16x32_f16; online softmax in exp2 domain.
// ---------------------------------------------------------------------------
__global__ __launch_bounds__(128) void fa_kernel(
    const _Float16* __restrict__ qh, const _Float16* __restrict__ kh,
    const _Float16* __restrict__ vT, float* __restrict__ ao) {
  __shared__ struct {
    _Float16 kbuf[2][32 * HD];   // 2 x 4KB : 32 keys x 64 d (row-major)
    _Float16 vbuf[2][HD * 32];   // 2 x 4KB : 64 d x 32 keys (row-major)
    _Float16 pbuf[4][16 * 32];   // 4KB     : per-wave P tile
  } smem;

  const int wave = threadIdx.x >> 5;
  const int lane = threadIdx.x & 31;
  const int lrow = lane & 15;
  const bool hi = lane >= 16;
  const int boff = hi ? 16 : 0;

  const int h = blockIdx.x >> 6;                 // 4 heads x 64 q-groups
  const int qblk = (blockIdx.x & 63) * 4 + wave; // all waves share head h

  // Q fragments (16 rows x 64 d, two K=32 chunks), already scaled+roped.
  const _Float16* qbase = qh + ((size_t)h * NQ + qblk * 16 + lrow) * HD;
  const half16 aq0 = load_a16(qbase, lane);
  const half16 aq1 = load_a16(qbase + 32, lane);

  f32x8 acc0 = {}, acc1 = {}, acc2 = {}, acc3 = {};
  float m[8], lsum[8];
#pragma unroll
  for (int r = 0; r < 8; ++r) { m[r] = -INFINITY; lsum[r] = 0.0f; }

  const _Float16* kbase_g = kh + (size_t)h * NK * HD;
  const _Float16* vbase_g = vT + (size_t)h * HD * NK;
  const int nblk = NK / 32;

  if (wave == 0) {  // prologue: DMA block 0 into buffer 0
    tdm_load_2d_f16(lds_off(smem.kbuf[0]), kbase_g, HD, 32, HD, NK, HD);
    tdm_load_2d_f16(lds_off(smem.vbuf[0]), vbase_g, 32, HD, NK, HD, NK);
  }

  for (int i = 0; i < nblk; ++i) {
    const int cur = i & 1, nxt = cur ^ 1;
    if (wave == 0 && i + 1 < nblk) {  // prefetch next block via TDM
      tdm_load_2d_f16(lds_off(smem.kbuf[nxt]),
                      kbase_g + (size_t)(i + 1) * 32 * HD, HD, 32, HD, NK, HD);
      tdm_load_2d_f16(lds_off(smem.vbuf[nxt]),
                      vbase_g + (size_t)(i + 1) * 32, 32, HD, NK, HD, NK);
    }
    if (i + 1 < nblk) __builtin_amdgcn_s_wait_tensorcnt(2);
    else              __builtin_amdgcn_s_wait_tensorcnt(0);
    __syncthreads();  // block i's K/V tiles visible to all waves

    // ---- scores: two 16x16 tiles (keys 0-15, 16-31 of this block) ----
    const _Float16* k0 = smem.kbuf[cur] + lrow * HD + boff;
    const _Float16* k1 = k0 + 16 * HD;
    f32x8 s0 = {}, s1 = {};
    s0 = wmma_f16(aq0, load_b16(k0), s0);
    s0 = wmma_f16(aq1, load_b16(k0 + 32), s0);
    s1 = wmma_f16(aq0, load_b16(k1), s1);
    s1 = wmma_f16(aq1, load_b16(k1 + 32), s1);

    // ---- online softmax (per row, 16-lane reductions within half-wave) ----
    _Float16* myP = smem.pbuf[wave];
#pragma unroll
    for (int r = 0; r < 8; ++r) {
      float x0 = s0[r], x1 = s1[r];
      float mx = fmaxf(x0, x1);
      mx = fmaxf(mx, __shfl_xor(mx, 1, 32));
      mx = fmaxf(mx, __shfl_xor(mx, 2, 32));
      mx = fmaxf(mx, __shfl_xor(mx, 4, 32));
      mx = fmaxf(mx, __shfl_xor(mx, 8, 32));
      const float mn = fmaxf(m[r], mx);
      const float sc = exp2f(m[r] - mn);
      const float p0 = exp2f(x0 - mn);
      const float p1 = exp2f(x1 - mn);
      float ps = p0 + p1;
      ps += __shfl_xor(ps, 1, 32);
      ps += __shfl_xor(ps, 2, 32);
      ps += __shfl_xor(ps, 4, 32);
      ps += __shfl_xor(ps, 8, 32);
      lsum[r] = lsum[r] * sc + ps;
      m[r] = mn;
      acc0[r] *= sc; acc1[r] *= sc; acc2[r] *= sc; acc3[r] *= sc;
      const int rowM = r + (hi ? 8 : 0);
      myP[rowM * 32 + lrow] = (_Float16)p0;
      myP[rowM * 32 + 16 + lrow] = (_Float16)p1;
    }
    asm volatile("s_wait_dscnt 0" ::: "memory");  // DS stores -> DS loads

    // P (16x32) C-layout -> A-layout via LDS
    const half16 aP = load_a16(myP + lrow * 32, lane);

    // ---- P @ V : four 16x16 d-chunks from the LDS V tile ----
    const _Float16* vb = smem.vbuf[cur] + lrow * 32 + boff;
    acc0 = wmma_f16(aP, load_b16(vb + 0 * 16 * 32), acc0);
    acc1 = wmma_f16(aP, load_b16(vb + 1 * 16 * 32), acc1);
    acc2 = wmma_f16(aP, load_b16(vb + 2 * 16 * 32), acc2);
    acc3 = wmma_f16(aP, load_b16(vb + 3 * 16 * 32), acc3);

    __syncthreads();  // everyone done with buffer `cur` before it is refilled
  }

  // ---- normalize and write AO[n][h*64+d] (f32) ----
#pragma unroll
  for (int r = 0; r < 8; ++r) {
    const int row = qblk * 16 + r + (hi ? 8 : 0);
    const float inv = 1.0f / lsum[r];
    float* dst = ao + (size_t)row * IDIM + h * HD + lrow;
    dst[0]  = acc0[r] * inv;
    dst[16] = acc1[r] * inv;
    dst[32] = acc2[r] * inv;
    dst[48] = acc3[r] * inv;
  }
}

extern "C" void kernel_launch(void* const* d_in, const int* in_sizes, int n_in,
                              void* d_out, int out_size, void* d_ws,
                              size_t ws_size, hipStream_t stream) {
  const float* query = (const float*)d_in[0];
  const float* key   = (const float*)d_in[1];
  const float* value = (const float*)d_in[2];
  const float* Wq = (const float*)d_in[3];
  const float* bq = (const float*)d_in[4];
  const float* Wk = (const float*)d_in[5];
  const float* bk = (const float*)d_in[6];
  const float* Wv = (const float*)d_in[7];
  const float* bv = (const float*)d_in[8];
  const float* Wo = (const float*)d_in[9];
  const float* bo = (const float*)d_in[10];
  const int* nke = (const int*)d_in[11];
  float* out = (float*)d_out;

  char* ws = (char*)d_ws;
  _Float16* qh = (_Float16*)(ws);                          // 4*4096*64*2 = 2MB
  _Float16* kh = (_Float16*)(ws + ((size_t)2 << 20));      // 4*8192*64*2 = 4MB
  _Float16* vT = (_Float16*)(ws + ((size_t)6 << 20));      // 4MB
  float*    ao = (float*)   (ws + ((size_t)10 << 20));     // 4096*256*4 = 4MB

  // Q/K/V projections (+bias, +rope, q-scale), f32 WMMA, epilogue to f16
  proj_kernel<<<(NQ / 16) * 4, 128, 0, stream>>>(query, Wq, bq, NQ, 256, 0, qh,
                                                 nullptr, nke);
  proj_kernel<<<(NK / 16) * 4, 128, 0, stream>>>(key, Wk, bk, NK, 64, 1, kh,
                                                 nullptr, nke);
  proj_kernel<<<(NK / 16) * 4, 128, 0, stream>>>(value, Wv, bv, NK, 64, 2, vT,
                                                 nullptr, nke);
  // attention core: f16 WMMA flash attention, TDM-fed K/V tiles in LDS
  fa_kernel<<<HEADS * (NQ / 16) / 4, 128, 0, stream>>>(qh, kh, vT, ao);
  // output projection, f32 WMMA, f32 result
  proj_kernel<<<(NQ / 16) * 4, 128, 0, stream>>>(ao, Wo, bo, NQ, 256, 3,
                                                 nullptr, out, nke);
}
